// EquivNoPhys_19713899889050
// MI455X (gfx1250) — matrix operand order repaired and checked by reading, not compiled, not run on hardware
//
#include <hip/hip_runtime.h>
#include <hip/hip_bf16.h>
#include <cstdint>
#include <cstddef>

// ---------------------------------------------------------------------------
// CDNA5 (gfx1250) EGNN+GIN pipeline.
// Every small dense layer (fan_in <= 33, fan_out = 16) runs as
// V_WMMA_F32_16X16X32_F16 with M = 16 edges/nodes per wave32; weights are
// pre-swizzled into the B-operand lane layout; activations transpose through
// LDS (in-order same-wave DS + s_wait_dscnt). Edge gathers use f16 node
// features (halves HBM gather traffic at 23.3 TB/s); segment sums use f32
// global atomics. All divisions in hot paths use v_rcp_f32 (one VALU op)
// instead of IEEE div expansion so VALU co-executes with the WMMA pipe.
// ---------------------------------------------------------------------------

typedef _Float16 h16;
typedef __attribute__((ext_vector_type(16))) _Float16 v16h;
typedef __attribute__((ext_vector_type(8)))  _Float16 v8h;
typedef __attribute__((ext_vector_type(8)))  float    v8f;

#define WPB 8   // waves per block (256 threads, wave32)

__device__ __forceinline__ float frcp(float x)   { return __builtin_amdgcn_rcpf(x); }
__device__ __forceinline__ float siluf(float x)  { return x * frcp(1.f + __expf(-x)); }
__device__ __forceinline__ float elu01f(float v) { return v > 0.f ? v : 0.1f * (__expf(v) - 1.f); }
__device__ __forceinline__ void lds_fence() { asm volatile("s_wait_dscnt 0" ::: "memory"); }

__device__ __forceinline__ v8f wmma16(v16h a, v16h b, v8f c) {
  // D = A(16x32 f16) * B(32x16 f16) + C(16x16 f32)
  return __builtin_amdgcn_wmma_f32_16x16x32_f16(false, a, false, b, (short)0, c,
                                                false, false);
}
__device__ __forceinline__ v16h load_btile(const h16* blob, int lane) {
  return *((const v16h*)blob + lane);   // 32B per lane, pre-swizzled
}

__device__ __forceinline__ void atomicMaxF(float* addr, float val) {
  unsigned int* ua = (unsigned int*)addr;
  unsigned int old = *ua;
  while (__uint_as_float(old) < val) {
    unsigned int assumed = old;
    old = atomicCAS(ua, assumed, __float_as_uint(val));
    if (old == assumed) break;
  }
}

// ---------------------------------------------------------------------------
// Weight pre-swizzle: B operand layout for v_wmma_f32_16x16x32_f16.
// lane l: column n = l&15, K-group g = l>>4 ; half h -> k = koff + (h<8 ? g*8+h : 16+g*8+h-8).
// Rows k >= fan_in are zero-filled (K padding).
// ---------------------------------------------------------------------------
struct PrepParams { const float* src[13]; int fan_in[13]; int koff[13]; };

__global__ void __launch_bounds__(256) k_prep(PrepParams p, h16* wt) {
  int idx = blockIdx.x * blockDim.x + threadIdx.x;
  if (idx >= 13 * 512) return;
  int tile = idx >> 9, w = idx & 511;
  int lane = w >> 4, h = w & 15;
  int n = lane & 15, g = lane >> 4;
  int k = p.koff[tile] + ((h < 8) ? (g * 8 + h) : (16 + g * 8 + (h - 8)));
  float v = (k < p.fan_in[tile]) ? p.src[tile][k * 16 + n] : 0.f;
  wt[tile * 512 + lane * 16 + h] = (h16)v;
}

__global__ void __launch_bounds__(256) k_cnt(const int* ei, float* cnt, int E) {
  int e = blockIdx.x * blockDim.x + threadIdx.x;
  if (e < E) atomicAdd(&cnt[ei[e]], 1.f);
}

// ---------------------------------------------------------------------------
// eq1 edge kernel: edge1 is 1->16 (scalar, produced directly in A layout),
// then WMMA edge2 (16->16), scatter messages; WMMA coord1, dot coord2,
// scatter coord deltas.
// ---------------------------------------------------------------------------
__global__ void __launch_bounds__(256) k_edge_eq1(
    const int* __restrict__ ei, const float* __restrict__ pos,
    const float* w_e1, const float* b_e1,
    const h16* wt_e2, const float* b_e2,
    const h16* wt_c1, const float* b_c1, const float* w_c2,
    float* __restrict__ agg, float* __restrict__ csum, int E)
{
  __shared__ h16   smh[WPB][16][16];
  __shared__ float smd[WPB][16][3];
  __shared__ int   smr[WPB][16];
  int lane = threadIdx.x & 31, wv = threadIdx.x >> 5;
  long tile = (long)blockIdx.x * WPB + wv;
  long e0 = tile * 16;
  if (e0 >= E) return;
  int m = lane & 15, g = lane >> 4, n = m;
  long e = e0 + m; if (e >= E) e = E - 1;
  int r = ei[e], c = ei[(long)E + e];
  float d0 = pos[r*3+0] - pos[c*3+0];
  float d1 = pos[r*3+1] - pos[c*3+1];
  float d2 = pos[r*3+2] - pos[c*3+2];
  float rad = d0*d0 + d1*d1 + d2*d2;
  if (g == 0) { smr[wv][m] = r; smd[wv][m][0] = d0; smd[wv][m][1] = d1; smd[wv][m][2] = d2; }

  // edge1 (1->16) directly in A layout for edge2 (effective K = 16)
  v16h a;
  #pragma unroll
  for (int h = 0; h < 16; h++) {
    if (h < 8) { int k = g*8 + h; a[h] = (h16)siluf(rad * w_e1[k] + b_e1[k]); }
    else a[h] = (h16)0.f;
  }
  v8f acc; float bb = b_e2[n];
  #pragma unroll
  for (int q = 0; q < 8; q++) acc[q] = bb;
  acc = wmma16(a, load_btile(wt_e2, lane), acc);
  #pragma unroll
  for (int q = 0; q < 8; q++) acc[q] = siluf(acc[q]);

  // stash messages for coord1 + scatter node aggregation
  #pragma unroll
  for (int q = 0; q < 8; q++) smh[wv][q + 8*g][n] = (h16)acc[q];
  lds_fence();
  #pragma unroll
  for (int q = 0; q < 8; q++) {
    long ee = e0 + q + 8*g;
    if (ee < E) { int rr = smr[wv][q + 8*g]; atomicAdd(&agg[(long)rr*16 + n], acc[q]); }
  }

  // coord1 (16->16)
  v16h a2;
  #pragma unroll
  for (int h = 0; h < 16; h++) {
    if (h < 8) a2[h] = smh[wv][m][g*8 + h]; else a2[h] = (h16)0.f;
  }
  v8f c1; float bc = b_c1[n];
  #pragma unroll
  for (int q = 0; q < 8; q++) c1[q] = bc;
  c1 = wmma16(a2, load_btile(wt_c1, lane), c1);
  #pragma unroll
  for (int q = 0; q < 8; q++) c1[q] = siluf(c1[q]);

  // coord2 (16->1): cross-lane dot, then scatter diff*phi
  float w2 = w_c2[n];
  #pragma unroll
  for (int q = 0; q < 8; q++) {
    float ph = c1[q] * w2;
    ph += __shfl_xor(ph, 1, 32); ph += __shfl_xor(ph, 2, 32);
    ph += __shfl_xor(ph, 4, 32); ph += __shfl_xor(ph, 8, 32);
    if (n < 3) {
      long ee = e0 + q + 8*g;
      if (ee < E) {
        int rr = smr[wv][q + 8*g];
        atomicAdd(&csum[(long)rr*3 + n], smd[wv][q + 8*g][n] * ph);
      }
    }
  }
}

// ---------------------------------------------------------------------------
// eq2 edge kernel: gather h[row],h[col] (f16, two b128 loads per lane),
// edge1 (33->16, K padded to 64 -> 2 WMMAs), edge2 (16->16), scatter agg.
// Coordinate path is dead after eq2 in the reference -> skipped.
// ---------------------------------------------------------------------------
__global__ void __launch_bounds__(256) k_edge_eq2(
    const int* __restrict__ ei, const float* __restrict__ pos,
    const h16* __restrict__ hf,
    const h16* wt_e1a, const h16* wt_e1b, const float* b_e1,
    const h16* wt_e2, const float* b_e2,
    float* __restrict__ agg, int E)
{
  __shared__ h16 smh[WPB][16][16];
  __shared__ int smr[WPB][16];
  int lane = threadIdx.x & 31, wv = threadIdx.x >> 5;
  long tile = (long)blockIdx.x * WPB + wv;
  long e0 = tile * 16;
  if (e0 >= E) return;
  int m = lane & 15, g = lane >> 4, n = m;
  long e = e0 + m; if (e >= E) e = E - 1;
  int r = ei[e], c = ei[(long)E + e];
  float d0 = pos[r*3+0] - pos[c*3+0];
  float d1 = pos[r*3+1] - pos[c*3+1];
  float d2 = pos[r*3+2] - pos[c*3+2];
  float rad = d0*d0 + d1*d1 + d2*d2;
  if (g == 0) smr[wv][m] = r;

  // A tile0: k<16 = h[row], k in [16,32) = h[col]
  v8h lo = *(const v8h*)(hf + (long)r*16 + g*8);
  v8h hi = *(const v8h*)(hf + (long)c*16 + g*8);
  v16h a0;
  #pragma unroll
  for (int h = 0; h < 16; h++) a0[h] = (h < 8) ? lo[h] : hi[h - 8];
  // A tile1: only k=32 (radial) nonzero -> lane group 0, half 0
  v16h a1;
  #pragma unroll
  for (int h = 0; h < 16; h++) a1[h] = (h16)0.f;
  a1[0] = (g == 0) ? (h16)rad : (h16)0.f;

  v8f acc; float bb = b_e1[n];
  #pragma unroll
  for (int q = 0; q < 8; q++) acc[q] = bb;
  acc = wmma16(a0, load_btile(wt_e1a, lane), acc);
  acc = wmma16(a1, load_btile(wt_e1b, lane), acc);
  #pragma unroll
  for (int q = 0; q < 8; q++) acc[q] = siluf(acc[q]);

  // transpose through LDS, then edge2 (16->16)
  #pragma unroll
  for (int q = 0; q < 8; q++) smh[wv][q + 8*g][n] = (h16)acc[q];
  lds_fence();
  v16h a2;
  #pragma unroll
  for (int h = 0; h < 16; h++) {
    if (h < 8) a2[h] = smh[wv][m][g*8 + h]; else a2[h] = (h16)0.f;
  }
  v8f acc2; float b2v = b_e2[n];
  #pragma unroll
  for (int q = 0; q < 8; q++) acc2[q] = b2v;
  acc2 = wmma16(a2, load_btile(wt_e2, lane), acc2);
  #pragma unroll
  for (int q = 0; q < 8; q++) {
    float mv = siluf(acc2[q]);
    long ee = e0 + q + 8*g;
    if (ee < E) { int rr = smr[wv][q + 8*g]; atomicAdd(&agg[(long)rr*16 + n], mv); }
  }
}

// ---------------------------------------------------------------------------
// eq1 node kernel: node1(agg 16->16) -> silu -> node2 (16->16); coord update.
// ---------------------------------------------------------------------------
__global__ void __launch_bounds__(256) k_node_eq1(
    const float* __restrict__ agg, const float* __restrict__ csum,
    const float* __restrict__ cnt, const float* __restrict__ x,
    const h16* wt_n1, const float* b_n1, const h16* wt_n2, const float* b_n2,
    float* __restrict__ pos1, float* __restrict__ h1f, h16* __restrict__ h1h, int N)
{
  __shared__ h16 smh[WPB][16][16];
  int lane = threadIdx.x & 31, wv = threadIdx.x >> 5;
  long tile = (long)blockIdx.x * WPB + wv;
  long i0 = tile * 16;
  if (i0 >= N) return;
  int m = lane & 15, g = lane >> 4, n = m;
  long i = i0 + m; if (i >= N) i = N - 1;

  v8f t = *(const v8f*)(agg + i*16 + g*8);
  v16h a;
  #pragma unroll
  for (int h = 0; h < 16; h++) a[h] = (h < 8) ? (h16)t[h] : (h16)0.f;
  v8f acc; float bb = b_n1[n];
  #pragma unroll
  for (int q = 0; q < 8; q++) acc[q] = bb;
  acc = wmma16(a, load_btile(wt_n1, lane), acc);
  #pragma unroll
  for (int q = 0; q < 8; q++) smh[wv][q + 8*g][n] = (h16)siluf(acc[q]);
  lds_fence();
  v16h a2;
  #pragma unroll
  for (int h = 0; h < 16; h++) {
    if (h < 8) a2[h] = smh[wv][m][g*8 + h]; else a2[h] = (h16)0.f;
  }
  v8f acc2; float b2v = b_n2[n];
  #pragma unroll
  for (int q = 0; q < 8; q++) acc2[q] = b2v;
  acc2 = wmma16(a2, load_btile(wt_n2, lane), acc2);
  #pragma unroll
  for (int q = 0; q < 8; q++) {
    long ni = i0 + q + 8*g;
    if (ni < N) { h1f[ni*16 + n] = acc2[q]; h1h[ni*16 + n] = (h16)acc2[q]; }
  }
  if (g == 0 && (i0 + m) < N) {
    float rc = frcp(fmaxf(cnt[i], 1.f));
    #pragma unroll
    for (int j = 0; j < 3; j++) pos1[i*3 + j] = x[i*3 + j] + csum[i*3 + j] * rc;
  }
}

// ---------------------------------------------------------------------------
// eq2 node kernel: node1(cat(h,agg) 32->16, full-K WMMA) -> silu -> node2,
// residual add.
// ---------------------------------------------------------------------------
__global__ void __launch_bounds__(256) k_node_eq2(
    const float* __restrict__ h1f, const h16* __restrict__ h1h,
    const float* __restrict__ agg,
    const h16* wt_n1, const float* b_n1, const h16* wt_n2, const float* b_n2,
    float* __restrict__ repF, h16* __restrict__ repH, int N)
{
  __shared__ h16 smh[WPB][16][16];
  int lane = threadIdx.x & 31, wv = threadIdx.x >> 5;
  long tile = (long)blockIdx.x * WPB + wv;
  long i0 = tile * 16;
  if (i0 >= N) return;
  int m = lane & 15, g = lane >> 4, n = m;
  long i = i0 + m; if (i >= N) i = N - 1;

  v8h lo = *(const v8h*)(h1h + i*16 + g*8);          // k < 16 : h
  v8f hiF = *(const v8f*)(agg + i*16 + g*8);         // k >= 16 : agg
  v16h a;
  #pragma unroll
  for (int h = 0; h < 16; h++) a[h] = (h < 8) ? lo[h] : (h16)hiF[h - 8];
  v8f acc; float bb = b_n1[n];
  #pragma unroll
  for (int q = 0; q < 8; q++) acc[q] = bb;
  acc = wmma16(a, load_btile(wt_n1, lane), acc);
  #pragma unroll
  for (int q = 0; q < 8; q++) smh[wv][q + 8*g][n] = (h16)siluf(acc[q]);
  lds_fence();
  v16h a2;
  #pragma unroll
  for (int h = 0; h < 16; h++) {
    if (h < 8) a2[h] = smh[wv][m][g*8 + h]; else a2[h] = (h16)0.f;
  }
  v8f acc2; float b2v = b_n2[n];
  #pragma unroll
  for (int q = 0; q < 8; q++) acc2[q] = b2v;
  acc2 = wmma16(a2, load_btile(wt_n2, lane), acc2);
  #pragma unroll
  for (int q = 0; q < 8; q++) {
    long ni = i0 + q + 8*g;
    if (ni < N) {
      float o = acc2[q] + h1f[ni*16 + n];   // residual
      repF[ni*16 + n] = o; repH[ni*16 + n] = (h16)o;
    }
  }
}

// GIN scatter: z[col] += hx[row] (f16 gather, f32 atomic accumulate)
__global__ void __launch_bounds__(256) k_gin_scatter(
    const int* __restrict__ ei, const h16* __restrict__ srcF,
    float* __restrict__ z, int E)
{
  long t = (long)blockIdx.x * blockDim.x + threadIdx.x;
  if (t >= (long)E * 16) return;
  long e = t >> 4; int f = (int)(t & 15);
  int r = ei[e], c = ei[(long)E + e];
  atomicAdd(&z[(long)c*16 + f], (float)srcF[(long)r*16 + f]);
}

// GIN lin1 + per-feature sum/sumsq for batch-norm
__global__ void __launch_bounds__(256) k_gin_lin1(
    const float* __restrict__ z, const h16* wt, const float* bias,
    float* __restrict__ h1g, float* __restrict__ stats, int N)
{
  int lane = threadIdx.x & 31, wv = threadIdx.x >> 5;
  long tile = (long)blockIdx.x * WPB + wv;
  long i0 = tile * 16;
  if (i0 >= N) return;
  int m = lane & 15, g = lane >> 4, n = m;
  long i = i0 + m; if (i >= N) i = N - 1;
  v8f t = *(const v8f*)(z + i*16 + g*8);
  v16h a;
  #pragma unroll
  for (int h = 0; h < 16; h++) a[h] = (h < 8) ? (h16)t[h] : (h16)0.f;
  v8f acc; float bb = bias[n];
  #pragma unroll
  for (int q = 0; q < 8; q++) acc[q] = bb;
  acc = wmma16(a, load_btile(wt, lane), acc);
  float s = 0.f, s2 = 0.f;
  #pragma unroll
  for (int q = 0; q < 8; q++) {
    long ni = i0 + q + 8*g;
    if (ni < N) { h1g[ni*16 + n] = acc[q]; s += acc[q]; s2 += acc[q]*acc[q]; }
  }
  s  += __shfl_xor(s, 16, 32);
  s2 += __shfl_xor(s2, 16, 32);
  if (lane < 16) { atomicAdd(&stats[n], s); atomicAdd(&stats[16 + n], s2); }
}

__global__ void k_gin_fin(float* stats, const float* gamma, const float* beta, float invN) {
  int n = threadIdx.x;
  if (n >= 16) return;
  float mu  = stats[n] * invN;
  float var = stats[16 + n] * invN - mu * mu;
  float inv = rsqrtf(var + 1e-5f);
  stats[32 + n] = mu; stats[48 + n] = gamma[n] * inv; stats[64 + n] = beta[n];
}

// GIN: normalize + relu + lin2 + elu01
__global__ void __launch_bounds__(256) k_gin_lin2(
    const float* __restrict__ h1g, const float* __restrict__ stats,
    const h16* wt, const float* bias,
    float* __restrict__ outF, h16* __restrict__ outH, int N)
{
  int lane = threadIdx.x & 31, wv = threadIdx.x >> 5;
  long tile = (long)blockIdx.x * WPB + wv;
  long i0 = tile * 16;
  if (i0 >= N) return;
  int m = lane & 15, g = lane >> 4, n = m;
  long i = i0 + m; if (i >= N) i = N - 1;
  v16h a;
  #pragma unroll
  for (int h = 0; h < 16; h++) {
    if (h < 8) {
      int k = g*8 + h;
      float v = (h1g[i*16 + k] - stats[32 + k]) * stats[48 + k] + stats[64 + k];
      a[h] = (h16)fmaxf(v, 0.f);
    } else a[h] = (h16)0.f;
  }
  v8f acc; float bb = bias[n];
  #pragma unroll
  for (int q = 0; q < 8; q++) acc[q] = bb;
  acc = wmma16(a, load_btile(wt, lane), acc);
  #pragma unroll
  for (int q = 0; q < 8; q++) {
    long ni = i0 + q + 8*g;
    if (ni < N) {
      float o = elu01f(acc[q]);
      outF[ni*16 + n] = o; outH[ni*16 + n] = (h16)o;
    }
  }
}

__global__ void __launch_bounds__(256) k_pool_init(float* psum, float* pmax, float* pcnt, int G) {
  int idx = blockIdx.x * blockDim.x + threadIdx.x;
  if (idx < G * 16) { psum[idx] = 0.f; pmax[idx] = -INFINITY; }
  if (idx < G) pcnt[idx] = 0.f;
}

__global__ void __launch_bounds__(256) k_pool(
    const float* __restrict__ rep, const int* __restrict__ batch,
    float* psum, float* pmax, float* pcnt, int N)
{
  long t = (long)blockIdx.x * blockDim.x + threadIdx.x;
  if (t >= (long)N * 16) return;
  long i = t >> 4; int f = (int)(t & 15);
  int gi = batch[i];
  float v = rep[t];
  atomicAdd(&psum[gi*16 + f], v);
  atomicMaxF(&pmax[gi*16 + f], v);
  if (f == 0) atomicAdd(&pcnt[gi], 1.f);
}

__global__ void k_cls(const float* psum, const float* pmax, const float* pcnt,
                      const float* seg, const float* W1, const float* b1,
                      const float* W2, const float* b2, float* out, int G)
{
  int gi = blockIdx.x * blockDim.x + threadIdx.x;
  if (gi >= G) return;
  float feat[33];
  float rc = frcp(fmaxf(pcnt[gi], 1.f));
  for (int f = 0; f < 16; f++) feat[f]      = psum[gi*16 + f] * rc;
  for (int f = 0; f < 16; f++) feat[16 + f] = pmax[gi*16 + f];
  feat[32] = seg[gi];
  float h[16];
  for (int j = 0; j < 16; j++) {
    float s = b1[j];
    for (int f = 0; f < 33; f++) s += feat[f] * W1[f*16 + j];
    h[j] = elu01f(s);
  }
  float l0 = b2[0], l1 = b2[1];
  for (int j = 0; j < 16; j++) { l0 += h[j] * W2[j*2]; l1 += h[j] * W2[j*2 + 1]; }
  float mx = fmaxf(l0, l1);
  float e0 = __expf(l0 - mx), e1 = __expf(l1 - mx);
  float rs = frcp(e0 + e1);
  out[gi*2] = e0 * rs; out[gi*2 + 1] = e1 * rs;
}

// ---------------------------------------------------------------------------
extern "C" void kernel_launch(void* const* d_in, const int* in_sizes, int n_in,
                              void* d_out, int out_size, void* d_ws, size_t ws_size,
                              hipStream_t stream) {
  (void)n_in; (void)out_size; (void)ws_size;
  // dict-tree-flatten order: batch, edge_index, params(sorted nested), segment, x
  const int*   batch  = (const int*)d_in[0];
  const int*   ei     = (const int*)d_in[1];
  const float* cls1W  = (const float*)d_in[2];
  const float* cls1b  = (const float*)d_in[3];
  const float* cls2W  = (const float*)d_in[4];
  const float* cls2b  = (const float*)d_in[5];
  // eq1: coord1{W,b}, coord2{W}, edge1{W,b}, edge2{W,b}, node1{W,b}, node2{W,b}
  const float* e1_c1W = (const float*)d_in[6];
  const float* e1_c1b = (const float*)d_in[7];
  const float* e1_c2W = (const float*)d_in[8];
  const float* e1_e1W = (const float*)d_in[9];
  const float* e1_e1b = (const float*)d_in[10];
  const float* e1_e2W = (const float*)d_in[11];
  const float* e1_e2b = (const float*)d_in[12];
  const float* e1_n1W = (const float*)d_in[13];
  const float* e1_n1b = (const float*)d_in[14];
  const float* e1_n2W = (const float*)d_in[15];
  const float* e1_n2b = (const float*)d_in[16];
  // eq2
  const float* e2_e1W = (const float*)d_in[20];
  const float* e2_e1b = (const float*)d_in[21];
  const float* e2_e2W = (const float*)d_in[22];
  const float* e2_e2b = (const float*)d_in[23];
  const float* e2_n1W = (const float*)d_in[24];
  const float* e2_n1b = (const float*)d_in[25];
  const float* e2_n2W = (const float*)d_in[26];
  const float* e2_n2b = (const float*)d_in[27];
  // gin1: beta, gamma, lin1{W,b}, lin2{W,b}
  const float* g1_beta  = (const float*)d_in[28];
  const float* g1_gamma = (const float*)d_in[29];
  const float* g1_l1W   = (const float*)d_in[30];
  const float* g1_l1b   = (const float*)d_in[31];
  const float* g1_l2W   = (const float*)d_in[32];
  const float* g1_l2b   = (const float*)d_in[33];
  const float* g2_beta  = (const float*)d_in[34];
  const float* g2_gamma = (const float*)d_in[35];
  const float* g2_l1W   = (const float*)d_in[36];
  const float* g2_l1b   = (const float*)d_in[37];
  const float* g2_l2W   = (const float*)d_in[38];
  const float* g2_l2b   = (const float*)d_in[39];
  const float* seg = (const float*)d_in[40];
  const float* x   = (const float*)d_in[41];

  int N = in_sizes[0];
  int E = in_sizes[1] / 2;
  int G = in_sizes[40];
  float* out = (float*)d_out;

  // workspace layout
  char* p = (char*)d_ws;
  auto alloc = [&](size_t b) -> void* {
    void* r = (void*)p; p += (b + 255) & ~(size_t)255; return r;
  };
  h16*   WT    = (h16*)alloc(13 * 512 * sizeof(h16));
  float* CNT   = (float*)alloc((size_t)N * 4);
  float* POS1  = (float*)alloc((size_t)N * 3 * 4);
  float* CSUM  = (float*)alloc((size_t)N * 3 * 4);
  float* AGGZ  = (float*)alloc((size_t)N * 16 * 4);
  float* H1F   = (float*)alloc((size_t)N * 16 * 4);
  h16*   H1H   = (h16*)alloc((size_t)N * 16 * 2);
  float* REPF  = (float*)alloc((size_t)N * 16 * 4);
  h16*   REPH  = (h16*)alloc((size_t)N * 16 * 2);
  float* STATS = (float*)alloc(80 * 4);
  float* PSUM  = (float*)alloc((size_t)G * 16 * 4);
  float* PMAX  = (float*)alloc((size_t)G * 16 * 4);
  float* PCNT  = (float*)alloc((size_t)G * 4);

  // pre-swizzle B tiles
  PrepParams pp;
  const float* srcs[13] = { e1_e2W, e1_c1W, e1_n1W, e1_n2W,
                            e2_e1W, e2_e1W, e2_e2W, e2_n1W, e2_n2W,
                            g1_l1W, g1_l2W, g2_l1W, g2_l2W };
  int fins[13]  = {16,16,16,16, 33,33,16,32,16, 16,16,16,16};
  int koffs[13] = { 0, 0, 0, 0,  0,32, 0, 0, 0,  0, 0, 0, 0};
  for (int t = 0; t < 13; t++) { pp.src[t] = srcs[t]; pp.fan_in[t] = fins[t]; pp.koff[t] = koffs[t]; }
  k_prep<<<(13*512 + 255)/256, 256, 0, stream>>>(pp, WT);

  hipMemsetAsync(CNT,  0, (size_t)N * 4, stream);
  hipMemsetAsync(CSUM, 0, (size_t)N * 3 * 4, stream);
  hipMemsetAsync(AGGZ, 0, (size_t)N * 16 * 4, stream);
  k_cnt<<<(E + 255)/256, 256, 0, stream>>>(ei, CNT, E);

  int eTiles = (E + 15) / 16, eBlocks = (eTiles + WPB - 1) / WPB;
  int nTiles = (N + 15) / 16, nBlocks = (nTiles + WPB - 1) / WPB;

  // ---- EGCL 1 ----
  k_edge_eq1<<<eBlocks, 256, 0, stream>>>(ei, x, e1_e1W, e1_e1b,
      WT + 0*512, e1_e2b, WT + 1*512, e1_c1b, e1_c2W, AGGZ, CSUM, E);
  k_node_eq1<<<nBlocks, 256, 0, stream>>>(AGGZ, CSUM, CNT, x,
      WT + 2*512, e1_n1b, WT + 3*512, e1_n2b, POS1, H1F, H1H, N);

  // ---- EGCL 2 (coord path dead after this layer) ----
  hipMemsetAsync(AGGZ, 0, (size_t)N * 16 * 4, stream);
  k_edge_eq2<<<eBlocks, 256, 0, stream>>>(ei, POS1, H1H,
      WT + 4*512, WT + 5*512, e2_e1b, WT + 6*512, e2_e2b, AGGZ, E);
  k_node_eq2<<<nBlocks, 256, 0, stream>>>(H1F, H1H, AGGZ,
      WT + 7*512, e2_n1b, WT + 8*512, e2_n2b, REPF, REPH, N);

  long st = (long)E * 16;
  int sBlocks = (int)((st + 255) / 256);

  // ---- GIN 1 ----
  hipMemcpyAsync(AGGZ, REPF, (size_t)N * 16 * 4, hipMemcpyDeviceToDevice, stream);
  k_gin_scatter<<<sBlocks, 256, 0, stream>>>(ei, REPH, AGGZ, E);
  hipMemsetAsync(STATS, 0, 32 * 4, stream);
  k_gin_lin1<<<nBlocks, 256, 0, stream>>>(AGGZ, WT + 9*512, g1_l1b, H1F, STATS, N);
  k_gin_fin<<<1, 16, 0, stream>>>(STATS, g1_gamma, g1_beta, 1.f / (float)N);
  k_gin_lin2<<<nBlocks, 256, 0, stream>>>(H1F, STATS, WT + 10*512, g1_l2b, REPF, REPH, N);

  // ---- GIN 2 ----
  hipMemcpyAsync(AGGZ, REPF, (size_t)N * 16 * 4, hipMemcpyDeviceToDevice, stream);
  k_gin_scatter<<<sBlocks, 256, 0, stream>>>(ei, REPH, AGGZ, E);
  hipMemsetAsync(STATS, 0, 32 * 4, stream);
  k_gin_lin1<<<nBlocks, 256, 0, stream>>>(AGGZ, WT + 11*512, g2_l1b, H1F, STATS, N);
  k_gin_fin<<<1, 16, 0, stream>>>(STATS, g2_gamma, g2_beta, 1.f / (float)N);
  k_gin_lin2<<<nBlocks, 256, 0, stream>>>(H1F, STATS, WT + 12*512, g2_l2b, REPF, REPH, N);

  // ---- pooling + classifier ----
  k_pool_init<<<(G*16 + 255)/256, 256, 0, stream>>>(PSUM, PMAX, PCNT, G);
  k_pool<<<(int)(((long)N*16 + 255)/256), 256, 0, stream>>>(REPF, batch, PSUM, PMAX, PCNT, N);
  k_cls<<<(G + 63)/64, 64, 0, stream>>>(PSUM, PMAX, PCNT, seg, cls1W, cls1b, cls2W, cls2b, out, G);
}
